// MoE_34651796144726
// MI455X (gfx1250) — matrix operand order repaired and checked by reading, not compiled, not run on hardware
//
#include <hip/hip_runtime.h>
#include <hip/hip_bf16.h>
#include <math.h>

// Problem sizes (match reference)
#define BTOK 2048
#define DM   768
#define NE   8
#define TOPK 2
#define HH   3072   // 4*DM

typedef _Float16 v16h __attribute__((ext_vector_type(16)));
typedef _Float16 v8h  __attribute__((ext_vector_type(8)));
typedef _Float16 h4v  __attribute__((ext_vector_type(4)));
typedef float    v8f  __attribute__((ext_vector_type(8)));
typedef float    f4v  __attribute__((ext_vector_type(4)));

union AFrag { v16h v; v8h h[2]; };

// ---------------------------------------------------------------- zero counters
__global__ void moe_zero_kernel(int* cnt) {
    if (threadIdx.x < NE) cnt[threadIdx.x] = 0;
}

// ---------------------------------------------------------------- f32 -> f16 weight conversion (x4 vectorized)
__global__ void moe_convert_kernel(const float* __restrict__ W1,
                                   const float* __restrict__ W2,
                                   _Float16* __restrict__ W1h,
                                   _Float16* __restrict__ W2h,
                                   size_t n4) {
    size_t i = (size_t)blockIdx.x * blockDim.x + threadIdx.x;
    size_t stride = (size_t)gridDim.x * blockDim.x;
    const f4v* w1 = (const f4v*)W1;
    const f4v* w2 = (const f4v*)W2;
    h4v* o1 = (h4v*)W1h;
    h4v* o2 = (h4v*)W2h;
    for (; i < n4; i += stride) {
        o1[i] = __builtin_convertvector(w1[i], h4v);
        o2[i] = __builtin_convertvector(w2[i], h4v);
    }
}

// ---------------------------------------------------------------- gating: logits, top-2, routing lists
__global__ __launch_bounds__(256) void moe_gating_kernel(
    const float* __restrict__ x, const float* __restrict__ wg,
    int* __restrict__ cnt, int* __restrict__ eslot, float* __restrict__ egate,
    int* __restrict__ ridx, float* __restrict__ g0a, float* __restrict__ g1a)
{
    __shared__ float wgs[DM * NE];
    for (int i = threadIdx.x; i < DM * NE; i += 256) wgs[i] = wg[i];
    __syncthreads();

    const int row = blockIdx.x * 256 + threadIdx.x;
    const float* xr = x + (size_t)row * DM;
    float acc[NE];
#pragma unroll
    for (int e = 0; e < NE; e++) acc[e] = 0.f;
    for (int d = 0; d < DM; d++) {
        float xv = xr[d];
#pragma unroll
        for (int e = 0; e < NE; e++) acc[e] = fmaf(xv, wgs[d * NE + e], acc[e]);
    }
    // top-2 (ties -> lower index, like lax.top_k)
    float v0 = -INFINITY, v1 = -INFINITY; int i0 = 0, i1 = 0;
#pragma unroll
    for (int e = 0; e < NE; e++) {
        float v = acc[e];
        if (v > v0) { v1 = v0; i1 = i0; v0 = v; i0 = e; }
        else if (v > v1) { v1 = v; i1 = e; }
    }
    // softmax over the two selected logits (v0 >= v1)
    float g0 = 1.f / (1.f + expf(v1 - v0));
    float g1 = 1.f - g0;

    int p0 = atomicAdd(&cnt[i0], 1);
    eslot[i0 * BTOK + p0] = row * 2;
    egate[i0 * BTOK + p0] = g0;
    int p1 = atomicAdd(&cnt[i1], 1);
    eslot[i1 * BTOK + p1] = row * 2 + 1;
    egate[i1 * BTOK + p1] = g1;

    ridx[row] = i0 | (i1 << 4);
    g0a[row] = g0;
    g1a[row] = g1;
}

// ---------------------------------------------------------------- fused expert FFN (grouped GEMM, f16 WMMA, f32 accum)
// Block = 256 threads = 8 waves; one block handles a tile of 16 tokens of one expert.
#define XLDSTRIDE 776   // 768 + 8 half pad -> 4-bank row offset, conflict-free b128 reads
#define HLDSTRIDE 136   // 128 + 8 half pad

__global__ __launch_bounds__(256) void moe_ffn_kernel(
    const float* __restrict__ x,
    const _Float16* __restrict__ W1h, const float* __restrict__ b1,
    const _Float16* __restrict__ W2h, const float* __restrict__ b2,
    const int* __restrict__ cnt, const int* __restrict__ eslot,
    const float* __restrict__ egate, float* __restrict__ contrib)
{
    __shared__ _Float16 xlds[16 * XLDSTRIDE];
    __shared__ _Float16 hlds[16 * HLDSTRIDE];
    __shared__ int   slot_s[16];
    __shared__ float gate_s[16];

    const int e = blockIdx.x >> 7;     // expert
    const int t = blockIdx.x & 127;    // tile slot within expert
    const int n_e = cnt[e];
    if (t * 16 >= n_e) return;

    const int tid  = threadIdx.x;
    const int lane = tid & 31;
    const int wave = tid >> 5;

    if (tid < 16) {
        int i = t * 16 + tid;
        if (i < n_e) { slot_s[tid] = eslot[e * BTOK + i]; gate_s[tid] = egate[e * BTOK + i]; }
        else         { slot_s[tid] = -1;                  gate_s[tid] = 0.f; }
    }
    __syncthreads();

    // gather 16 token rows of x -> f16 LDS (16 threads per row, coalesced in d)
    {
        int r  = tid >> 4;
        int c0 = tid & 15;
        int s  = slot_s[r];
        int row = (s >= 0) ? (s >> 1) : 0;
        const float* xr = x + (size_t)row * DM;
        for (int d = c0; d < DM; d += 16) xlds[r * XLDSTRIDE + d] = (_Float16)xr[d];
    }
    __syncthreads();

    const int hl = lane & 15;            // M (or N) index within half-wave
    const int kq = (lane >> 4) << 3;     // K sub-offset: 0 or 8 (A-fragment layout)
    const int mb = (lane >> 4) << 3;     // C/D row base: 0 or 8

    v8f acc2[6];
#pragma unroll
    for (int i = 0; i < 6; i++) acc2[i] = (v8f){0.f,0.f,0.f,0.f,0.f,0.f,0.f,0.f};

    const _Float16* W1e = W1h + (size_t)e * DM * HH;
    const _Float16* W2e = W2h + (size_t)e * HH * DM;

    for (int hb = 0; hb < HH; hb += 128) {
        // ---- GEMM1: h[16 x 16] for cols [hb + wave*16) ; K = 768
        v8f acc1 = (v8f){0.f,0.f,0.f,0.f,0.f,0.f,0.f,0.f};
        const _Float16* w1col = W1e + hb + wave * 16;
        for (int k0 = 0; k0 < DM; k0 += 32) {
            AFrag a;
            a.h[0] = *(const v8h*)&xlds[hl * XLDSTRIDE + k0 + kq];
            a.h[1] = *(const v8h*)&xlds[hl * XLDSTRIDE + k0 + kq + 16];
            // B fragment: column N = hl, K in same order as A fragment
            v16h bv;
#pragma unroll
            for (int s = 0; s < 16; s++) bv[s] = w1col[(size_t)(k0 + kq + (s < 8 ? s : s + 8)) * HH + hl];
            acc1 = __builtin_amdgcn_wmma_f32_16x16x32_f16(
                false, a.v, false, bv, (short)0, acc1, false, false);
        }
        // bias + exact GELU -> f16 into hlds
        {
            int col = wave * 16 + hl;
            float bias = b1[e * HH + hb + col];
#pragma unroll
            for (int r = 0; r < 8; r++) {
                float v = acc1[r] + bias;
                float g = 0.5f * v * (1.0f + erff(v * 0.70710678118654752f));
                hlds[(mb + r) * HLDSTRIDE + col] = (_Float16)g;
            }
        }
        __syncthreads();

        // ---- GEMM2: accumulate out[16 x 768] += h_chunk[16 x 128] @ W2[hb:hb+128, :]
#pragma unroll
        for (int i = 0; i < 6; i++) {
            int nt = wave * 6 + i;                      // 48 N-tiles over 8 waves
            const _Float16* w2col = W2e + (size_t)hb * DM + nt * 16;
            v8f c = acc2[i];
#pragma unroll
            for (int k0 = 0; k0 < 128; k0 += 32) {
                AFrag a;
                a.h[0] = *(const v8h*)&hlds[hl * HLDSTRIDE + k0 + kq];
                a.h[1] = *(const v8h*)&hlds[hl * HLDSTRIDE + k0 + kq + 16];
                v16h bv;
#pragma unroll
                for (int s = 0; s < 16; s++) bv[s] = w2col[(size_t)(k0 + kq + (s < 8 ? s : s + 8)) * DM + hl];
                c = __builtin_amdgcn_wmma_f32_16x16x32_f16(
                    false, a.v, false, bv, (short)0, c, false, false);
            }
            acc2[i] = c;
        }
        __syncthreads();
    }

    // ---- epilogue: +b2, x gate, scatter to contrib[slot][d] (each slot written once)
    {
#pragma unroll
        for (int i = 0; i < 6; i++) {
            int nt = wave * 6 + i;
            int d  = nt * 16 + hl;
            float bias = b2[e * DM + d];
#pragma unroll
            for (int r = 0; r < 8; r++) {
                int m = mb + r;
                int s = slot_s[m];
                if (s >= 0) {
                    contrib[(size_t)s * DM + d] = (acc2[i][r] + bias) * gate_s[m];
                }
            }
        }
    }
}

// ---------------------------------------------------------------- combine: y[b] = contrib[2b] + contrib[2b+1]
__global__ void moe_combine_kernel(const float* __restrict__ contrib, float* __restrict__ y) {
    size_t i = (size_t)blockIdx.x * 256 + threadIdx.x;
    if (i >= (size_t)BTOK * DM) return;
    size_t b = i / DM, d = i % DM;
    y[i] = contrib[(2 * b) * DM + d] + contrib[(2 * b + 1) * DM + d];
}

// ---------------------------------------------------------------- loss: deterministic single-thread reduction
__global__ void moe_loss_kernel(const int* __restrict__ ridx,
                                const float* __restrict__ g0a,
                                const float* __restrict__ g1a,
                                float* __restrict__ out) {
    if (threadIdx.x != 0 || blockIdx.x != 0) return;
    float imp[NE], ld[NE];
    for (int e = 0; e < NE; e++) { imp[e] = 0.f; ld[e] = 0.f; }
    for (int b = 0; b < BTOK; b++) {
        int ii = ridx[b];
        int e0 = ii & 15, e1 = ii >> 4;
        float g0 = g0a[b], g1 = g1a[b];
        imp[e0] += g0; if (g0 > 0.f) ld[e0] += 1.f;
        imp[e1] += g1; if (g1 > 0.f) ld[e1] += 1.f;
    }
    float loss = 0.f;
    for (int pass = 0; pass < 2; pass++) {
        const float* v = pass ? ld : imp;
        float m = 0.f;
        for (int e = 0; e < NE; e++) m += v[e];
        m /= (float)NE;
        float var = 0.f;
        for (int e = 0; e < NE; e++) { float dd = v[e] - m; var += dd * dd; }
        var /= (float)(NE - 1);                       // unbiased (ddof=1)
        loss += var / (m * m + 1e-10f);
    }
    out[(size_t)BTOK * DM] = loss * 1e-2f;
}

// ---------------------------------------------------------------- launcher
extern "C" void kernel_launch(void* const* d_in, const int* in_sizes, int n_in,
                              void* d_out, int out_size, void* d_ws, size_t ws_size,
                              hipStream_t stream) {
    (void)in_sizes; (void)n_in; (void)out_size; (void)ws_size;
    const float* x  = (const float*)d_in[0];
    const float* wg = (const float*)d_in[1];
    const float* W1 = (const float*)d_in[2];
    const float* b1 = (const float*)d_in[3];
    const float* W2 = (const float*)d_in[4];
    const float* b2 = (const float*)d_in[5];
    float* y = (float*)d_out;

    const size_t EDH = (size_t)NE * DM * HH;          // 18,874,368
    char* ws = (char*)d_ws;
    _Float16* W1h    = (_Float16*)(ws);
    _Float16* W2h    = (_Float16*)(ws + 2 * EDH);     // + 37,748,736 B
    float*    contrib= (float*)   (ws + 4 * EDH);     // B*K*D*4 = 12,582,912 B
    char*     tail   = ws + 4 * EDH + (size_t)BTOK * TOPK * DM * 4;
    int*      cnt    = (int*)  (tail);
    int*      eslot  = (int*)  (tail + 256);
    float*    egate  = (float*)(tail + 256 + (size_t)NE * BTOK * 4);
    int*      ridx   = (int*)  (tail + 256 + 2 * (size_t)NE * BTOK * 4);
    float*    g0a    = (float*)(tail + 256 + 2 * (size_t)NE * BTOK * 4 + BTOK * 4);
    float*    g1a    = (float*)(tail + 256 + 2 * (size_t)NE * BTOK * 4 + 2 * BTOK * 4);

    moe_zero_kernel<<<1, 32, 0, stream>>>(cnt);
    moe_convert_kernel<<<4096, 256, 0, stream>>>(W1, W2, W1h, W2h, EDH / 4);
    moe_gating_kernel<<<BTOK / 256, 256, 0, stream>>>(x, wg, cnt, eslot, egate, ridx, g0a, g1a);
    moe_ffn_kernel<<<NE * (BTOK / 16), 256, 0, stream>>>(x, W1h, b1, W2h, b2, cnt, eslot, egate, contrib);
    moe_combine_kernel<<<((size_t)BTOK * DM + 255) / 256, 256, 0, stream>>>(contrib, y);
    moe_loss_kernel<<<1, 1, 0, stream>>>(ridx, g0a, g1a, y);
}
